// EncoderLayer_41480794145041
// MI455X (gfx1250) — matrix operand order repaired and checked by reading, not compiled, not run on hardware
//
#include <hip/hip_runtime.h>
#include <hip/hip_bf16.h>
#include <stdint.h>

typedef __attribute__((ext_vector_type(16))) __bf16 bf16x16;
typedef __attribute__((ext_vector_type(8)))  float  f32x8;
typedef __attribute__((ext_vector_type(4)))  unsigned int u32x4;
typedef __attribute__((ext_vector_type(8)))  int i32x8;
typedef __attribute__((ext_vector_type(4)))  int i32x4;

#define EMBED 1024
#define SEQ   1024
#define NBATCH 8
#define HEADS 16
#define HDIM  64
#define NTOK  (NBATCH * SEQ)   // 8192

#if defined(__has_builtin)
#if __has_builtin(__builtin_amdgcn_tensor_load_to_lds) && __has_builtin(__builtin_amdgcn_s_wait_tensorcnt)
#define USE_TDM 1
#endif
#endif
#ifndef USE_TDM
#define USE_TDM 0
#endif

static __device__ __forceinline__ __bf16 f2bf(float f) {
    uint32_t u = __builtin_bit_cast(uint32_t, f);
    uint32_t r = u + 0x7FFFu + ((u >> 16) & 1u);   // round-to-nearest-even
    unsigned short h = (unsigned short)(r >> 16);
    return __builtin_bit_cast(__bf16, h);
}

static __device__ __forceinline__ unsigned rfl(unsigned v) {
    return (unsigned)__builtin_amdgcn_readfirstlane((int)v);
}

union FragB { bf16x16 v; u32x4 q[2]; };

// ---------------------------------------------------------------------------
// TDM: DMA a 2-D bf16 tile (tileK inner elems x tileRows rows, row stride
// strideK elems) from global into LDS, with LDS row padding so ds_load_b128
// stays 16B-aligned and bank-spread.
//   GEMM tiles:  64B rows + 32B pad  -> padI=3 (16dw), padA=7 (8dw) -> 96B stride
//   Attn tiles: 128B rows + 32B pad  -> padI=4 (32dw), padA=7 (8dw) -> 160B stride
// D# layout per CDNA5 ISA. count=1, type=2.
// ---------------------------------------------------------------------------
#if USE_TDM
static __device__ __forceinline__ void tdm_load_2d(unsigned lds_addr, const void* gptr,
                                                   unsigned tensorDim0, unsigned tensorDim1,
                                                   unsigned tileK, unsigned tileRows,
                                                   unsigned strideK,
                                                   unsigned padI, unsigned padA) {
    const unsigned long long ga = (unsigned long long)gptr;
    u32x4 g0;
    g0[0] = 1u;                                              // count=1 (valid), no gather
    g0[1] = rfl(lds_addr);                                   // lds_addr
    g0[2] = rfl((unsigned)(ga & 0xFFFFFFFFu));               // global_addr[31:0]
    g0[3] = rfl((unsigned)((ga >> 32) & 0x01FFFFFFu)) | (2u << 30);  // addr[56:32] | type=2
    i32x8 g1;
    g1[0] = (int)((1u << 16) | (1u << 20) | (padI << 22) | (padA << 25)); // 2B data + pad
    g1[1] = (int)((tensorDim0 & 0xFFFFu) << 16);
    g1[2] = (int)(((tensorDim0 >> 16) & 0xFFFFu) | ((tensorDim1 & 0xFFFFu) << 16));
    g1[3] = (int)(((tensorDim1 >> 16) & 0xFFFFu) | ((tileK & 0xFFFFu) << 16)); // tile_dim0
    g1[4] = (int)(tileRows & 0xFFFFu);                       // tile_dim1, tile_dim2=0
    g1[5] = (int)strideK;                                    // tensor_dim0_stride[31:0]
    g1[6] = 0;
    g1[7] = 0;
    const i32x4 z4 = {0, 0, 0, 0};                           // groups 2/3 unused (2-D)
    const i32x8 z8 = {0, 0, 0, 0, 0, 0, 0, 0};
    __builtin_amdgcn_tensor_load_to_lds(g0, g1, z4, z4, z8, 0);
}
#endif

// ---------------------------------------------------------------------------
// fp32 [K][N] weight -> bf16 [N][K] transposed copy (once; GEMM B tiles then
// become plain 2-D tiles the TDM can DMA with zero in-kernel shuffling).
// ---------------------------------------------------------------------------
__global__ void __launch_bounds__(256) cvt_transpose_bf16(const float* __restrict__ in,
                                                          __bf16* __restrict__ out) {
    __shared__ float tile[32][33];
    const int k0 = blockIdx.x * 32;
    const int n0 = blockIdx.y * 32;
    const int c  = threadIdx.x & 31;
    const int r0 = threadIdx.x >> 5;   // 0..7
#pragma unroll
    for (int i = 0; i < 4; ++i) {
        const int r = r0 + i * 8;
        tile[r][c] = in[(size_t)(k0 + r) * EMBED + n0 + c];
    }
    __syncthreads();
#pragma unroll
    for (int i = 0; i < 4; ++i) {
        const int r = r0 + i * 8;
        out[(size_t)(n0 + r) * EMBED + k0 + c] = f2bf(tile[c][r]);
    }
}

// ---------------------------------------------------------------------------
// V [tok][E] bf16 -> per-head transposed vT[(b*H+h)*64 + d][SEQ] bf16, once.
// Moves the V transpose out of the (16x re-staged) attention inner loop and
// turns the attention V tile into a plain 2-D tile for the TDM.
// ---------------------------------------------------------------------------
__global__ void __launch_bounds__(256) transpose_v_heads(const __bf16* __restrict__ v,
                                                         __bf16* __restrict__ vt) {
    __shared__ __attribute__((aligned(16))) __bf16 t[64][72];  // [s][d], 144B rows
    const int bh = blockIdx.x;          // b*H + h
    const int b  = bh >> 4;
    const int h  = bh & 15;
    const int s0 = blockIdx.y * 64;
    const int tid = threadIdx.x;
#pragma unroll
    for (int i = 0; i < 2; ++i) {
        const int cid = tid + i * 256;
        const int r = cid >> 3;
        const int c = (cid & 7) * 8;
        *(u32x4*)&t[r][c] =
            *(const u32x4*)(v + (size_t)(b * SEQ + s0 + r) * EMBED + h * HDIM + c);
    }
    __syncthreads();
#pragma unroll
    for (int i = 0; i < 2; ++i) {
        const int cid = tid + i * 256;
        const int d  = cid >> 3;
        const int sc = (cid & 7) * 8;
        __bf16 tmp[8];
#pragma unroll
        for (int u = 0; u < 8; ++u) tmp[u] = t[sc + u][d];
        *(u32x4*)(vt + ((size_t)bh * HDIM + d) * SEQ + s0 + sc) = *(u32x4*)tmp;
    }
}

// ---------------------------------------------------------------------------
// LayerNorm (no affine) over last dim E=1024, output bf16. One row per block.
// ---------------------------------------------------------------------------
__global__ void __launch_bounds__(256) ln_rows(const float* __restrict__ x,
                                               __bf16* __restrict__ y) {
    __shared__ float rs[256];
    __shared__ float rq[256];
    const int row = blockIdx.x;
    const float* xr = x + (size_t)row * EMBED;
    float v0[4];
    float s = 0.f, q = 0.f;
#pragma unroll
    for (int i = 0; i < 4; ++i) {
        float v = xr[threadIdx.x + i * 256];
        v0[i] = v; s += v; q += v * v;
    }
    rs[threadIdx.x] = s; rq[threadIdx.x] = q;
    __syncthreads();
    for (int off = 128; off > 0; off >>= 1) {
        if (threadIdx.x < off) {
            rs[threadIdx.x] += rs[threadIdx.x + off];
            rq[threadIdx.x] += rq[threadIdx.x + off];
        }
        __syncthreads();
    }
    const float mean = rs[0] * (1.0f / EMBED);
    const float var  = rq[0] * (1.0f / EMBED) - mean * mean;
    const float inv  = rsqrtf(var + 1e-5f);
    __bf16* yr = y + (size_t)row * EMBED;
#pragma unroll
    for (int i = 0; i < 4; ++i)
        yr[threadIdx.x + i * 256] = f2bf((v0[i] - mean) * inv);
}

// ---------------------------------------------------------------------------
// WMMA bf16 GEMM: C = (A @ Wt^T + bias) * oscale (+ReLU / +residual).
// Block 256 thr (8 waves); tile 64(M) x 256(N); K step 32, double-buffered.
// Wave (wm 0..1, wn 0..3) owns a 32x64 slab: 8 WMMA / 12 ds_load_b128 per step.
// All global->LDS staging via Tensor Data Mover issued by wave 0.
// ---------------------------------------------------------------------------
template <bool RELU, bool RESID, bool OUTB>
__global__ void __launch_bounds__(256) gemm_wmma_tdm(
    const __bf16* __restrict__ A,     // M x K row-major bf16
    const __bf16* __restrict__ Wt,    // N x K row-major bf16 (pre-transposed)
    const float*  __restrict__ bias,  // N
    const float*  __restrict__ resid, // M x N fp32 (RESID)
    float*        __restrict__ outf,  // M x N fp32 (RESID)
    __bf16*       __restrict__ outb,  // M x N bf16 (OUTB)
    int M, int N, int K, float oscale)
{
    __shared__ __attribute__((aligned(16))) __bf16 As[2][64 * 48];
    __shared__ __attribute__((aligned(16))) __bf16 Bt[2][256 * 48];

    const int tid  = threadIdx.x;
    const int lane = tid & 31;
    const int wid  = tid >> 5;
    const int wm   = wid & 1;     // 0..1, 32 rows each
    const int wn   = wid >> 1;    // 0..3, 64 cols each
    const int mBase = blockIdx.x * 64;
    const int nBase = blockIdx.y * 256;

    const int kbA = (lane < 16) ? 0 : 8;    // A-frag K base (ISA 16-bit A 16x32)
    const int kbB = (lane < 16) ? 0 : 8;    // B-frag K base (ISA 16-bit B 32x16)

    f32x8 acc[2][4] = {};
    const int iters = K / 32;

#if USE_TDM
    if (wid == 0) {
        tdm_load_2d((unsigned)(size_t)&As[0][0], A  + (size_t)mBase * K, K, M, 32, 64,  K, 3, 7);
        tdm_load_2d((unsigned)(size_t)&Bt[0][0], Wt + (size_t)nBase * K, K, N, 32, 256, K, 3, 7);
    }
#endif

    for (int i = 0; i < iters; ++i) {
        const int buf = i & 1;
#if USE_TDM
        if (wid == 0) {
            if (i + 1 < iters) {
                const int k0 = (i + 1) * 32;
                tdm_load_2d((unsigned)(size_t)&As[buf ^ 1][0],
                            A + (size_t)mBase * K + k0, K, M, 32, 64, K, 3, 7);
                tdm_load_2d((unsigned)(size_t)&Bt[buf ^ 1][0],
                            Wt + (size_t)nBase * K + k0, K, N, 32, 256, K, 3, 7);
                __builtin_amdgcn_s_wait_tensorcnt(2);   // tile i pair complete
            } else {
                __builtin_amdgcn_s_wait_tensorcnt(0);
            }
        }
        __syncthreads();   // publish tile i; readers of buf^1 already past prev barrier
#else
        __syncthreads();
        {   // fallback: cooperative vector staging
            const int k0 = i * 32;
            const int ar = tid >> 2, ac = (tid & 3) * 8;
            *(u32x4*)(As[buf] + ar * 48 + ac) =
                *(const u32x4*)(A + (size_t)(mBase + ar) * K + k0 + ac);
#pragma unroll
            for (int p = 0; p < 4; ++p) {
                const int cid = tid + p * 256;
                const int nr = cid >> 2, nc = (cid & 3) * 8;
                *(u32x4*)(Bt[buf] + nr * 48 + nc) =
                    *(const u32x4*)(Wt + (size_t)(nBase + nr) * K + k0 + nc);
            }
        }
        __syncthreads();
#endif
        const __bf16* Ab = As[buf];
        const __bf16* Bb = Bt[buf];

        FragB af[2];
#pragma unroll
        for (int mt = 0; mt < 2; ++mt) {
            const int r = wm * 32 + mt * 16 + (lane & 15);
            af[mt].q[0] = *(const u32x4*)(Ab + r * 48 + kbA);
            af[mt].q[1] = *(const u32x4*)(Ab + r * 48 + kbA + 16);
        }
#pragma unroll
        for (int t = 0; t < 4; ++t) {
            FragB bw;
            const int nr = wn * 64 + t * 16 + (lane & 15);
            bw.q[0] = *(const u32x4*)(Bb + nr * 48 + kbB);
            bw.q[1] = *(const u32x4*)(Bb + nr * 48 + kbB + 16);
#pragma unroll
            for (int mt = 0; mt < 2; ++mt)
                acc[mt][t] = __builtin_amdgcn_wmma_f32_16x16x32_bf16(
                    false, af[mt].v, false, bw.v, (short)0, acc[mt][t], false, false);
        }
        __syncthreads();   // all reads of buf done before TDM refills it next iter
    }

    // epilogue (fully specialized; C-layout rows m = j + (lane>=16 ? 8 : 0))
#pragma unroll
    for (int mt = 0; mt < 2; ++mt) {
        const int rowBase = mBase + wm * 32 + mt * 16 + ((lane < 16) ? 0 : 8);
#pragma unroll
        for (int t = 0; t < 4; ++t) {
            const int col = nBase + wn * 64 + t * 16 + (lane & 15);
            const float bcol = bias[col];
#pragma unroll
            for (int j = 0; j < 8; ++j) {
                float v = (acc[mt][t][j] + bcol) * oscale;
                if (RELU) v = fmaxf(v, 0.f);
                const size_t idx = (size_t)(rowBase + j) * N + col;
                if (RESID) outf[idx] = v + resid[idx];
                if (OUTB)  outb[idx] = f2bf(v);
            }
        }
    }
}

// ---------------------------------------------------------------------------
// Flash-style attention, WMMA bf16, TDM-fed K/V tiles (double-buffered).
// Grid: (B*H, SEQ/64). Block: 128 thr (4 waves); wave owns 16 query rows.
// Q is pre-scaled by 1/8 in its projection GEMM.
// ---------------------------------------------------------------------------
__global__ void __launch_bounds__(128) attn_wmma(
    const __bf16* __restrict__ Q,    // [tok][E], pre-scaled by 0.125
    const __bf16* __restrict__ Kg,   // [tok][E]
    const __bf16* __restrict__ Vt,   // [(b*H+h)*64 + d][SEQ]
    __bf16* __restrict__ O)          // [tok][E]
{
    __shared__ __attribute__((aligned(16))) __bf16 Qs[64 * 80];
    __shared__ __attribute__((aligned(16))) __bf16 Ks[2][64 * 80];   // [key][d]
    __shared__ __attribute__((aligned(16))) __bf16 Vts[2][64 * 80];  // [d][key]
    __shared__ __attribute__((aligned(16))) __bf16 Ps[64 * 80];

    const int tid  = threadIdx.x;
    const int lane = tid & 31;
    const int w    = tid >> 5;
    const int bh   = blockIdx.x;
    const int b    = bh >> 4;
    const int h    = bh & 15;
    const int qb   = blockIdx.y;

    const size_t tokBase = (size_t)b * SEQ;
    const int colBase = h * HDIM;
    const __bf16* vhead = Vt + (size_t)bh * HDIM * SEQ;

    // stage Q tile (64 x 64 bf16), once
#pragma unroll
    for (int i = 0; i < 4; ++i) {
        const int cid = tid + i * 128;
        const int r = cid >> 3;
        const int c = (cid & 7) * 8;
        *(u32x4*)(Qs + r * 80 + c) =
            *(const u32x4*)(Q + (tokBase + qb * 64 + r) * EMBED + colBase + c);
    }
    __syncthreads();

    const int qrow = w * 16 + (lane & 15);
    const int kbA  = (lane < 16) ? 0 : 8;
    const int kbB  = (lane < 16) ? 0 : 8;

    FragB qa[2];
#pragma unroll
    for (int c = 0; c < 2; ++c) {
        qa[c].q[0] = *(const u32x4*)(Qs + qrow * 80 + c * 32 + kbA);
        qa[c].q[1] = *(const u32x4*)(Qs + qrow * 80 + c * 32 + kbA + 16);
    }

    float mrow[8], lrow[8];
    f32x8 o[4] = {};
#pragma unroll
    for (int j = 0; j < 8; ++j) { mrow[j] = -1e30f; lrow[j] = 0.f; }

    const int nblk = SEQ / 64;
#if USE_TDM
    if (w == 0) {
        tdm_load_2d((unsigned)(size_t)&Ks[0][0],  Kg + tokBase * EMBED + colBase,
                    EMBED, SEQ, 64, 64, EMBED, 4, 7);
        tdm_load_2d((unsigned)(size_t)&Vts[0][0], vhead,
                    SEQ, HDIM, 64, 64, SEQ, 4, 7);
    }
#endif

    for (int kblk = 0; kblk < nblk; ++kblk) {
        const int buf = kblk & 1;
#if USE_TDM
        if (w == 0) {
            if (kblk + 1 < nblk) {
                const int nk = (kblk + 1) * 64;
                tdm_load_2d((unsigned)(size_t)&Ks[buf ^ 1][0],
                            Kg + (tokBase + nk) * EMBED + colBase,
                            EMBED, SEQ, 64, 64, EMBED, 4, 7);
                tdm_load_2d((unsigned)(size_t)&Vts[buf ^ 1][0],
                            vhead + nk, SEQ, HDIM, 64, 64, SEQ, 4, 7);
                __builtin_amdgcn_s_wait_tensorcnt(2);
            } else {
                __builtin_amdgcn_s_wait_tensorcnt(0);
            }
        }
        __syncthreads();
#else
        __syncthreads();
#pragma unroll
        for (int i = 0; i < 4; ++i) {   // fallback vector staging (both row copies)
            const int cid = tid + i * 128;
            const int r = cid >> 3;
            const int c = (cid & 7) * 8;
            *(u32x4*)(Ks[buf] + r * 80 + c) =
                *(const u32x4*)(Kg + (tokBase + kblk * 64 + r) * EMBED + colBase + c);
            *(u32x4*)(Vts[buf] + r * 80 + c) =
                *(const u32x4*)(vhead + (size_t)r * SEQ + kblk * 64 + c);
        }
        __syncthreads();
#endif
        const __bf16* Kb = Ks[buf];
        const __bf16* Vb = Vts[buf];

        // S = Qs @ K^T (Q pre-scaled): 4 key tiles x (2 x K32)
        f32x8 sc[4] = {};
#pragma unroll
        for (int t = 0; t < 4; ++t) {
#pragma unroll
            for (int c = 0; c < 2; ++c) {
                FragB bk;
                const int keyrow = t * 16 + (lane & 15);
                const int doff = c * 32 + kbB;
                bk.q[0] = *(const u32x4*)(Kb + keyrow * 80 + doff);
                bk.q[1] = *(const u32x4*)(Kb + keyrow * 80 + doff + 16);
                sc[t] = __builtin_amdgcn_wmma_f32_16x16x32_bf16(
                    false, qa[c].v, false, bk.v, (short)0, sc[t], false, false);
            }
        }

        // online softmax; row reductions across 16-lane halves
#pragma unroll
        for (int j = 0; j < 8; ++j) {
            float mx = fmaxf(fmaxf(sc[0][j], sc[1][j]), fmaxf(sc[2][j], sc[3][j]));
#pragma unroll
            for (int m = 8; m >= 1; m >>= 1)
                mx = fmaxf(mx, __shfl_xor(mx, m, 16));
            const float mnew  = fmaxf(mrow[j], mx);
            const float scale = __expf(mrow[j] - mnew);
            float rsum = 0.f;
#pragma unroll
            for (int t = 0; t < 4; ++t) {
                const float p = __expf(sc[t][j] - mnew);
                sc[t][j] = p;
                rsum += p;
            }
#pragma unroll
            for (int m = 8; m >= 1; m >>= 1)
                rsum += __shfl_xor(rsum, m, 16);
            lrow[j] = lrow[j] * scale + rsum;
            mrow[j] = mnew;
#pragma unroll
            for (int t = 0; t < 4; ++t) o[t][j] = o[t][j] * scale;
        }

        // spill probs (C layout) to LDS, re-read in A-fragment layout (same wave)
        const int prBase = w * 16 + ((lane < 16) ? 0 : 8);
#pragma unroll
        for (int t = 0; t < 4; ++t)
#pragma unroll
            for (int j = 0; j < 8; ++j)
                Ps[(prBase + j) * 80 + t * 16 + (lane & 15)] = f2bf(sc[t][j]);

        // O += P @ V
#pragma unroll
        for (int c = 0; c < 2; ++c) {
            FragB pa;
            pa.q[0] = *(const u32x4*)(Ps + qrow * 80 + c * 32 + kbA);
            pa.q[1] = *(const u32x4*)(Ps + qrow * 80 + c * 32 + kbA + 16);
#pragma unroll
            for (int t = 0; t < 4; ++t) {
                FragB bvf;
                const int drow = t * 16 + (lane & 15);
                const int koff = c * 32 + kbB;
                bvf.q[0] = *(const u32x4*)(Vb + drow * 80 + koff);
                bvf.q[1] = *(const u32x4*)(Vb + drow * 80 + koff + 16);
                o[t] = __builtin_amdgcn_wmma_f32_16x16x32_bf16(
                    false, pa.v, false, bvf.v, (short)0, o[t], false, false);
            }
        }
        __syncthreads();   // readers of buf done before TDM refills it
    }

    const int orBase = w * 16 + ((lane < 16) ? 0 : 8);
#pragma unroll
    for (int t = 0; t < 4; ++t)
#pragma unroll
        for (int j = 0; j < 8; ++j) {
            const float v = o[t][j] / lrow[j];
            const size_t tok = tokBase + qb * 64 + orBase + j;
            O[tok * EMBED + colBase + t * 16 + (lane & 15)] = f2bf(v);
        }
}

// ---------------------------------------------------------------------------
extern "C" void kernel_launch(void* const* d_in, const int* in_sizes, int n_in,
                              void* d_out, int out_size, void* d_ws, size_t ws_size,
                              hipStream_t stream) {
    const float* x  = (const float*)d_in[0];
    const float* Wq = (const float*)d_in[1];  const float* bq = (const float*)d_in[2];
    const float* Wk = (const float*)d_in[3];  const float* bk = (const float*)d_in[4];
    const float* Wv = (const float*)d_in[5];  const float* bv = (const float*)d_in[6];
    const float* Wo = (const float*)d_in[7];  const float* bo = (const float*)d_in[8];
    const float* W1 = (const float*)d_in[9];  const float* b1 = (const float*)d_in[10];
    const float* W2 = (const float*)d_in[11]; const float* b2 = (const float*)d_in[12];
    float* out = (float*)d_out;

    char* ws = (char*)d_ws;
    size_t off = 0;
    auto alloc = [&](size_t bytes) -> void* {
        void* p = (void*)(ws + off);
        off += (bytes + 255) & ~(size_t)255;
        return p;
    };
    const size_t MTOK = NTOK, E = EMBED;
    __bf16* xn   = (__bf16*)alloc(MTOK * E * 2);
    __bf16* wqt  = (__bf16*)alloc(E * E * 2);
    __bf16* wkt  = (__bf16*)alloc(E * E * 2);
    __bf16* wvt  = (__bf16*)alloc(E * E * 2);
    __bf16* wot  = (__bf16*)alloc(E * E * 2);
    __bf16* w1t  = (__bf16*)alloc(E * E * 2);
    __bf16* w2t  = (__bf16*)alloc(E * E * 2);
    __bf16* qbuf = (__bf16*)alloc(MTOK * E * 2);
    __bf16* kbuf = (__bf16*)alloc(MTOK * E * 2);
    __bf16* vbuf = (__bf16*)alloc(MTOK * E * 2);
    __bf16* vT   = (__bf16*)alloc(MTOK * E * 2);
    __bf16* attn = (__bf16*)alloc(MTOK * E * 2);
    float*  hbuf = (float*)alloc(MTOK * E * 4);
    __bf16* hn   = (__bf16*)alloc(MTOK * E * 2);
    __bf16* m1   = (__bf16*)alloc(MTOK * E * 2);

    dim3 tg(32, 32);
    cvt_transpose_bf16<<<tg, 256, 0, stream>>>(Wq, wqt);
    cvt_transpose_bf16<<<tg, 256, 0, stream>>>(Wk, wkt);
    cvt_transpose_bf16<<<tg, 256, 0, stream>>>(Wv, wvt);
    cvt_transpose_bf16<<<tg, 256, 0, stream>>>(Wo, wot);
    cvt_transpose_bf16<<<tg, 256, 0, stream>>>(W1, w1t);
    cvt_transpose_bf16<<<tg, 256, 0, stream>>>(W2, w2t);

    ln_rows<<<NTOK, 256, 0, stream>>>(x, xn);

    dim3 gg(NTOK / 64, EMBED / 256);   // (128, 4)
    // Q pre-scaled by softmax 1/DIVIDER = 1/8
    gemm_wmma_tdm<false, false, true><<<gg, 256, 0, stream>>>(
        xn, wqt, bq, nullptr, nullptr, qbuf, NTOK, EMBED, EMBED, 0.125f);
    gemm_wmma_tdm<false, false, true><<<gg, 256, 0, stream>>>(
        xn, wkt, bk, nullptr, nullptr, kbuf, NTOK, EMBED, EMBED, 1.0f);
    gemm_wmma_tdm<false, false, true><<<gg, 256, 0, stream>>>(
        xn, wvt, bv, nullptr, nullptr, vbuf, NTOK, EMBED, EMBED, 1.0f);

    dim3 vg(NBATCH * HEADS, SEQ / 64);
    transpose_v_heads<<<vg, 256, 0, stream>>>(vbuf, vT);

    dim3 ag(NBATCH * HEADS, SEQ / 64);
    attn_wmma<<<ag, 128, 0, stream>>>(qbuf, kbuf, vT, attn);

    // h = x + attn @ Wo + bo
    gemm_wmma_tdm<false, true, false><<<gg, 256, 0, stream>>>(
        attn, wot, bo, x, hbuf, nullptr, NTOK, EMBED, EMBED, 1.0f);
    ln_rows<<<NTOK, 256, 0, stream>>>(hbuf, hn);
    // m1 = relu(hn @ W1 + b1)
    gemm_wmma_tdm<true, false, true><<<gg, 256, 0, stream>>>(
        hn, w1t, b1, nullptr, nullptr, m1, NTOK, EMBED, EMBED, 1.0f);
    // out = h + m1 @ W2 + b2
    gemm_wmma_tdm<false, true, false><<<gg, 256, 0, stream>>>(
        m1, w2t, b2, hbuf, out, nullptr, NTOK, EMBED, EMBED, 1.0f);
}
